// BSN_76218489635087
// MI455X (gfx1250) — matrix operand order repaired and checked by reading, not compiled, not run on hardware
//
#include <hip/hip_runtime.h>
#include <hip/hip_bf16.h>

typedef __attribute__((ext_vector_type(16))) _Float16 v16h;
typedef __attribute__((ext_vector_type(8)))  _Float16 v8h;
typedef __attribute__((ext_vector_type(8)))  float    v8f;

#define WMMA_F16(a, b, c) \
  __builtin_amdgcn_wmma_f32_16x16x32_f16(false, (a), false, (b), (short)0, (c), false, false)

#define CAT16 0,1,2,3,4,5,6,7,8,9,10,11,12,13,14,15

// ---------------- f32 -> f16 convert ----------------
__global__ void cvt_f32_f16(const float* __restrict__ src, _Float16* __restrict__ dst, int n) {
  int i = blockIdx.x * blockDim.x + threadIdx.x;
  if (i < n) dst[i] = (_Float16)src[i];
}

// ---------------- pack W [Kin,Kout] f32 into WMMA B-fragment layout, f16 ----------------
// For tile (kt, tn), lane l, element e: K = kt*32 + (l>>4)*16 + e, col = tn*16 + (l&15),
// stored at ((kt*tilesN + tn)*32 + l)*16 + e  => two contiguous b128 loads per lane in GEMM.
__global__ void pack_w_frag(const float* __restrict__ W, _Float16* __restrict__ Wp,
                            int Kout, int tilesN, int n) {
  int i = blockIdx.x * blockDim.x + threadIdx.x;
  if (i >= n) return;
  int e    = i & 15;
  int l    = (i >> 4) & 31;
  int tile = i >> 9;
  int tn   = tile % tilesN;
  int kt   = tile / tilesN;
  int K    = kt * 32 + (l >> 4) * 16 + e;
  int col  = tn * 16 + (l & 15);
  Wp[i] = (_Float16)W[(size_t)K * Kout + col];
}

// ---------------- fused GEMM + bias + ReLU (f16 in, f16 out, f32 accum) ----------------
__global__ void gemm_relu_f16(const _Float16* __restrict__ A, const _Float16* __restrict__ Wp,
                              const float* __restrict__ bias, _Float16* __restrict__ C,
                              int Kin, int Kout) {
  const int lane = threadIdx.x & 31;
  const int wave = threadIdx.x >> 5;
  const int tile = blockIdx.x * 4 + wave;       // 4 waves / block, grid sized exactly
  const int tilesN = Kout >> 4;
  const int tm = tile / tilesN;
  const int tn = tile % tilesN;
  const int hf = lane >> 4;
  const int m  = lane & 15;

  v8f c = {};
  const _Float16* arow = A + (size_t)(tm * 16 + m) * Kin + hf * 16;
  const _Float16* wlane = Wp + ((size_t)tn * 32 + lane) * 16;
  const size_t wstep = (size_t)tilesN * 32 * 16;
  for (int k0 = 0; k0 < Kin; k0 += 32) {
    v8h alo = *(const v8h*)(arow + k0);
    v8h ahi = *(const v8h*)(arow + k0 + 8);
    v16h a = __builtin_shufflevector(alo, ahi, CAT16);
    v8h blo = *(const v8h*)(wlane);
    v8h bhi = *(const v8h*)(wlane + 8);
    v16h b = __builtin_shufflevector(blo, bhi, CAT16);
    wlane += wstep;
    c = WMMA_F16(a, b, c);
  }
  const float bv = bias[tn * 16 + m];
  _Float16* crow = C + (size_t)(tm * 16 + hf * 8) * Kout + tn * 16 + m;
#pragma unroll
  for (int r = 0; r < 8; ++r) {
    float v = c[r] + bv;
    v = v > 0.f ? v : 0.f;
    crow[(size_t)r * Kout] = (_Float16)v;
  }
}

// ---------------- ordered-uint mapping for float atomic max ----------------
__device__ __forceinline__ unsigned f32_ord(float f) {
  unsigned u = __float_as_uint(f);
  return (u & 0x80000000u) ? ~u : (u | 0x80000000u);
}

__global__ void init_seg(unsigned* __restrict__ seg_ord) {
  int i = threadIdx.x;
  if (i < 100) seg_ord[i] = 0u;
}

// ---------------- fused score GEMM + max over N + segment atomic-max ----------------
// Register-blocked: each wave owns FOUR 16-column tiles (64 cols). B fragments
// (8 x v16h) are loop-invariant; each A-fragment load feeds 8 WMMAs, cutting the
// dominant L2 A-traffic 4x (1.6 GB -> 0.4 GB). s[2048,100000] never materialized.
// Tail handled branchlessly (clamped loads, guarded atomics) so EXEC stays all-1
// for every WMMA as the ISA requires.
__global__ void score_max4(const _Float16* __restrict__ H, const float* __restrict__ TR,
                           const int* __restrict__ mask, unsigned* __restrict__ seg_ord,
                           int T) {
  const int lane = threadIdx.x & 31;
  const int wave = threadIdx.x >> 5;
  const int g = blockIdx.x * 8 + wave;          // group of 4 col-tiles
  const int tc0 = g * 4;
  if (tc0 * 16 >= T) return;                    // wave-uniform guard
  const int hf = lane >> 4;
  const int m  = lane & 15;
  const int kb = hf * 16;

  v16h b[4][2];
  int cols[4];
#pragma unroll
  for (int t = 0; t < 4; ++t) {
    const int col = (tc0 + t) * 16 + m;
    cols[t] = col;
    const int cc = col < T ? col : T - 1;       // clamp: safe OOB load, result discarded
    const float* tp = TR + cc;
#pragma unroll
    for (int e = 0; e < 16; ++e) b[t][0][e] = (_Float16)tp[(size_t)(kb + e) * T];
#pragma unroll
    for (int e = 0; e < 16; ++e) b[t][1][e] = (_Float16)tp[(size_t)(32 + kb + e) * T];
  }

  float mx[4] = {-3.4e38f, -3.4e38f, -3.4e38f, -3.4e38f};
  const _Float16* hp = H + (size_t)m * 64 + hf * 16;
  for (int nt = 0; nt < 2048 / 16; ++nt) {
    v8h a00 = *(const v8h*)(hp);
    v8h a01 = *(const v8h*)(hp + 8);
    v8h a10 = *(const v8h*)(hp + 32);
    v8h a11 = *(const v8h*)(hp + 40);
    v16h a0 = __builtin_shufflevector(a00, a01, CAT16);
    v16h a1 = __builtin_shufflevector(a10, a11, CAT16);
#pragma unroll
    for (int t = 0; t < 4; ++t) {
      v8f c = {};
      c = WMMA_F16(a0, b[t][0], c);
      c = WMMA_F16(a1, b[t][1], c);
#pragma unroll
      for (int r = 0; r < 8; ++r) mx[t] = fmaxf(mx[t], c[r]);
    }
    hp += 16 * 64;
  }
#pragma unroll
  for (int t = 0; t < 4; ++t) {
    float v = fmaxf(mx[t], __shfl_xor(mx[t], 16, 32));   // rows 0-7 vs 8-15 (wave32)
    if (hf == 0 && cols[t] < T) {
      int seg = mask[cols[t]];
      if (seg >= 0 && seg < 100) atomicMax(&seg_ord[seg], f32_ord(v));
    }
  }
}

// ---------------- final 100-dot + sigmoid + threshold ----------------
__global__ void finalize(const unsigned* __restrict__ seg_ord, const float* __restrict__ W4,
                         const float* __restrict__ b4, float* __restrict__ out) {
  int lane = threadIdx.x & 31;
  float s = 0.f;
  for (int r = lane; r < 100; r += 32) {
    unsigned key = seg_ord[r];
    unsigned u = (key & 0x80000000u) ? (key & 0x7FFFFFFFu) : ~key;
    s += __uint_as_float(u) * W4[r];
  }
#pragma unroll
  for (int off = 16; off; off >>= 1) s += __shfl_xor(s, off, 32);
  if (lane == 0) {
    float z = s + b4[0];
    float p = 1.f / (1.f + __expf(-z));
    out[0] = p;
    out[1] = (p >= 0.5f) ? 1.f : 0.f;
  }
}

// ---------------- host-side orchestration ----------------
extern "C" void kernel_launch(void* const* d_in, const int* in_sizes, int n_in,
                              void* d_out, int out_size, void* d_ws, size_t ws_size,
                              hipStream_t stream) {
  (void)in_sizes; (void)n_in; (void)out_size; (void)ws_size;
  const float* x   = (const float*)d_in[0];   // [1,2048,512]
  const float* tr  = (const float*)d_in[1];   // [64,100000]
  const int*   msk = (const int*)d_in[2];     // [100000]
  const float* W1  = (const float*)d_in[3];   // [512,256]
  const float* b1  = (const float*)d_in[4];
  const float* W2  = (const float*)d_in[5];   // [256,128]
  const float* b2  = (const float*)d_in[6];
  const float* W3  = (const float*)d_in[7];   // [128,64]
  const float* b3  = (const float*)d_in[8];
  const float* W4  = (const float*)d_in[9];   // [100,1]
  const float* b4  = (const float*)d_in[10];
  float* out = (float*)d_out;

  const int N = 2048, D = 512, H1 = 256, H2 = 128, H3 = 64, T = 100000;

  char* ws = (char*)d_ws;
  size_t off = 0;
  auto carve = [&](size_t bytes) -> void* {
    void* p = ws + off;
    off += (bytes + 255) & ~(size_t)255;
    return p;
  };
  _Float16* xh  = (_Float16*)carve((size_t)N * D * 2);
  _Float16* W1p = (_Float16*)carve((size_t)D * H1 * 2);
  _Float16* W2p = (_Float16*)carve((size_t)H1 * H2 * 2);
  _Float16* W3p = (_Float16*)carve((size_t)H2 * H3 * 2);
  _Float16* h1  = (_Float16*)carve((size_t)N * H1 * 2);
  _Float16* h2  = (_Float16*)carve((size_t)N * H2 * 2);
  _Float16* h3  = (_Float16*)carve((size_t)N * H3 * 2);
  unsigned* seg = (unsigned*)carve(100 * sizeof(unsigned));

  // 1) convert x to f16; pack weights into WMMA B-fragment layout (f16)
  cvt_f32_f16<<<(N * D + 255) / 256, 256, 0, stream>>>(x, xh, N * D);
  pack_w_frag<<<(D * H1 + 255) / 256, 256, 0, stream>>>(W1, W1p, H1, H1 / 16, D * H1);
  pack_w_frag<<<(H1 * H2 + 255) / 256, 256, 0, stream>>>(W2, W2p, H2, H2 / 16, H1 * H2);
  pack_w_frag<<<(H2 * H3 + 255) / 256, 256, 0, stream>>>(W3, W3p, H3, H3 / 16, H2 * H3);

  // 2) MLP: three WMMA GEMM+ReLU layers
  gemm_relu_f16<<<(N / 16) * (H1 / 16) / 4, 128, 0, stream>>>(xh, W1p, b1, h1, D, H1);
  gemm_relu_f16<<<(N / 16) * (H2 / 16) / 4, 128, 0, stream>>>(h1, W2p, b2, h2, H1, H2);
  gemm_relu_f16<<<(N / 16) * (H3 / 16) / 4, 128, 0, stream>>>(h2, W3p, b3, h3, H2, H3);

  // 3) fused score GEMM + row-max + segment-max; 4 col-tiles per wave
  init_seg<<<1, 128, 0, stream>>>(seg);
  {
    int groups = (T / 16 + 3) / 4;              // 1563
    int blocks = (groups + 7) / 8;              // 8 waves / block
    score_max4<<<blocks, 256, 0, stream>>>(h3, tr, msk, seg, T);
  }

  // 4) ref_max . W4 + b4 -> sigmoid -> (prob, hard)
  finalize<<<1, 32, 0, stream>>>(seg, W4, b4, out);
}
